// QuantLinear_15539191677564
// MI455X (gfx1250) — matrix operand order repaired and checked by reading, not compiled, not run on hardware
//
#include <hip/hip_runtime.h>

// ---------------------------------------------------------------------------
// NF4 quantized linear:  out[M,N] = x[M,K] * dequant(qweight)[K,N] + bias[N]
//   M = 8192 (4*2048), K = 4096, N = 4096, group size 128 along K.
// Fused dequant + f16 WMMA GEMM, f32 accumulation. Block tile 128x128, BK=64.
// 256 threads (8 wave32), each wave computes a 64x32 sub-tile (4x2 WMMA tiles).
// ---------------------------------------------------------------------------

typedef __attribute__((ext_vector_type(16))) _Float16 v16h;
typedef __attribute__((ext_vector_type(8)))  _Float16 v8h;
typedef __attribute__((ext_vector_type(4)))  _Float16 v4h;
typedef __attribute__((ext_vector_type(8)))  float    v8f;

#define BM 128
#define BN 128
#define BK 64
#define LDA 72   // padded row stride (halves) for As  [m][k]
#define LDB 72   // padded row stride (halves) for Bs  [n][k]

__constant__ float NF4_TAB[16] = {
    -1.0f, -0.6961928009986877f, -0.5250730514526367f, -0.39491748809814453f,
    -0.28444138169288635f, -0.18477343022823334f, -0.09105003625154495f, 0.0f,
    0.07958029955625534f, 0.16093020141124725f, 0.24611230194568634f,
    0.33791524171829224f, 0.44070982933044434f, 0.5626170039176941f,
    0.7229568362236023f, 1.0f};

__global__ __launch_bounds__(256)
void nf4_wmma_gemm(const float* __restrict__ x,
                   const float* __restrict__ scales,
                   const float* __restrict__ bias,
                   const int*   __restrict__ qw,
                   float* __restrict__ out,
                   int M, int K, int N)
{
    __shared__ _Float16 As[BM * LDA];   // 18.0 KB  (f16 activations)
    __shared__ _Float16 Bs[BN * LDB];   // 18.0 KB  (f16 dequantized weights, [n][k])
    __shared__ float    tab[16];        // NF4 code -> float

    const int tid = threadIdx.x;
    if (tid < 16) tab[tid] = NF4_TAB[tid];

    const int gm = blockIdx.y * BM;
    const int gn = blockIdx.x * BN;

    const int lane = tid & 31;
    const int w    = tid >> 5;          // wave id 0..7
    const int wm   = (w >> 2) * 64;     // wave M offset within block tile
    const int wn   = (w & 3)  * 32;     // wave N offset within block tile
    const int lm   = lane & 15;         // lane within half-wave
    const int lhi  = lane >> 4;         // 0: lanes 0-15, 1: lanes 16-31

    v8f acc[4][2];
#pragma unroll
    for (int mt = 0; mt < 4; ++mt)
#pragma unroll
        for (int nt = 0; nt < 2; ++nt)
            acc[mt][nt] = (v8f){0.f,0.f,0.f,0.f,0.f,0.f,0.f,0.f};

    __syncthreads();   // tab ready

    for (int k0 = 0; k0 < K; k0 += BK) {
        // ---- prefetch next x chunk into caches (global_prefetch_b8) ----
        if (k0 + BK < K) {
            const int prow = tid >> 1;                 // 128 rows / 2 points
            __builtin_prefetch(&x[(size_t)(gm + prow) * K + (k0 + BK) + ((tid & 1) << 5)], 0, 3);
        }

        // ---- stage A: x tile 128x64 f32 -> f16 in LDS ----
#pragma unroll
        for (int i = 0; i < 8; ++i) {
            const int idx = tid + i * 256;             // 0..2047 -> 2048 float4
            const int row = idx >> 4;                  // 0..127
            const int c4  = (idx & 15) << 2;           // k offset 0..60 step 4
            const float4 f = *reinterpret_cast<const float4*>(
                &x[(size_t)(gm + row) * K + k0 + c4]);
            v4h h;
            h[0] = (_Float16)f.x; h[1] = (_Float16)f.y;
            h[2] = (_Float16)f.z; h[3] = (_Float16)f.w;
            *reinterpret_cast<v4h*>(&As[row * LDA + c4]) = h;
        }

        // ---- stage B: decode NF4 tile 64(K)x128(N) -> f16 in LDS [n][k] ----
        const int krow0 = k0 >> 3;                     // qweight row (8 codes/int32)
        const int grp   = k0 >> 7;                     // scale group (BK=64 < 128)
#pragma unroll
        for (int i = 0; i < 4; ++i) {
            const int idx = tid + i * 256;             // 0..1023 int32 words
            const int nl  = idx & 127;                 // column within tile
            const int kr  = idx >> 7;                  // 0..7 packed-K row
            const unsigned int q =
                (unsigned int)qw[(size_t)(krow0 + kr) * N + gn + nl];
            const float s = scales[(size_t)grp * N + gn + nl];
            v8h hv;
#pragma unroll
            for (int j = 0; j < 8; ++j)
                hv[j] = (_Float16)(tab[(q >> (4 * j)) & 15u] * s);
            *reinterpret_cast<v8h*>(&Bs[nl * LDB + (kr << 3)]) = hv;
        }

        __syncthreads();

        // ---- WMMA over this K-chunk: two K=32 steps ----
#pragma unroll
        for (int kk = 0; kk < BK; kk += 32) {
            v16h af[4];
#pragma unroll
            for (int mt = 0; mt < 4; ++mt) {
                // A 16x32 f16 layout: lanes 0-15 hold K 0-7 / 16-23,
                // lanes 16-31 hold K 8-15 / 24-31 (ISA 7.12.2)
                const int base = (wm + mt * 16 + lm) * LDA + kk + (lhi << 3);
                const v8h lo = *reinterpret_cast<const v8h*>(&As[base]);
                const v8h hi = *reinterpret_cast<const v8h*>(&As[base + 16]);
#pragma unroll
                for (int j = 0; j < 8; ++j) { af[mt][j] = lo[j]; af[mt][j + 8] = hi[j]; }
            }
            v16h bf[2];
#pragma unroll
            for (int nt = 0; nt < 2; ++nt) {
                // B 32x16: lanes 0-15 hold K 0-7 / 16-23, lanes 16-31 hold K 8-15 / 24-31, N = lane&15
                const int base = (wn + nt * 16 + lm) * LDB + kk + (lhi << 3);
                const v8h lo = *reinterpret_cast<const v8h*>(&Bs[base]);
                const v8h hi = *reinterpret_cast<const v8h*>(&Bs[base + 16]);
#pragma unroll
                for (int j = 0; j < 8; ++j) { bf[nt][j] = lo[j]; bf[nt][j + 8] = hi[j]; }
            }
#pragma unroll
            for (int mt = 0; mt < 4; ++mt)
#pragma unroll
                for (int nt = 0; nt < 2; ++nt)
                    acc[mt][nt] = __builtin_amdgcn_wmma_f32_16x16x32_f16(
                        false, af[mt], false, bf[nt],
                        (short)0, acc[mt][nt], false, false);
        }

        __syncthreads();
    }

    // ---- epilogue: C/D layout M = r + 8*lhi, N = lane&15 ----
#pragma unroll
    for (int nt = 0; nt < 2; ++nt) {
        const int ng = gn + wn + nt * 16 + lm;
        const float bv = bias[ng];
#pragma unroll
        for (int mt = 0; mt < 4; ++mt) {
            const int mbase = gm + wm + mt * 16 + (lhi << 3);
#pragma unroll
            for (int r = 0; r < 8; ++r)
                out[(size_t)(mbase + r) * N + ng] = acc[mt][nt][r] + bv;
        }
    }
}

extern "C" void kernel_launch(void* const* d_in, const int* in_sizes, int n_in,
                              void* d_out, int out_size, void* d_ws, size_t ws_size,
                              hipStream_t stream) {
    const float* x      = (const float*)d_in[0];
    const float* scales = (const float*)d_in[1];
    const float* bias   = (const float*)d_in[2];
    const int*   qw     = (const int*)d_in[3];
    float*       out    = (float*)d_out;

    const int N = in_sizes[2];                                   // OUT = 4096
    const int K = (int)(((long long)in_sizes[3] * 8) / N);       // IN  = 4096
    const int M = (int)((long long)in_sizes[0] / K);             // 8192

    dim3 grid(N / BN, M / BM);
    nf4_wmma_gemm<<<grid, 256, 0, stream>>>(x, scales, bias, qw, out, M, K, N);
}
